// DeformableConv1D_3899830304947
// MI455X (gfx1250) — matrix operand order repaired and checked by reading, not compiled, not run on hardware
//
#include <hip/hip_runtime.h>
#include <hip/hip_bf16.h>
#include <math.h>

// ---------------------------------------------------------------------------
// DeformableConv1D for MI455X (gfx1250).
// B=8, C=256, L=4096, O_CH=256, K=3, OG=4, HID=64, GROUPS=1, L_OUT=4096.
// Main einsum = GEMM M=256, N=32768, Kred=768 (12.9 GFLOP) -> WMMA bf16.
// x (33.5MB) is L2-resident (192MB L2); HBM floor ~67MB => memory-bound.
// Strategy: gather+lerp once in f32, convert to bf16, stage B tiles in LDS in
// the native WMMA operand layout (ping-pong double buffered, 1 barrier/chunk),
// pre-repack weights into the native A layout. N-tile = 32 per block so each
// A fetch feeds 2 WMMAs; 4 WMMAs per wave per barrier.
// ---------------------------------------------------------------------------

#define LSEQ   4096
#define NCH    256
#define NB     8
#define KTAP   3
#define NCHUNK 24   // 768 / 32 reduction chunks

typedef __attribute__((ext_vector_type(16))) __bf16 v16bf;
typedef __attribute__((ext_vector_type(8)))  float  v8f;

__device__ __forceinline__ unsigned short f32_to_bf16_rne(float f) {
    unsigned u = __builtin_bit_cast(unsigned, f);
    unsigned r = (u + 0x7FFFu + ((u >> 16) & 1u)) >> 16;
    return (unsigned short)r;
}

// ---------------------------------------------------------------------------
// Kernel 1: offset/modulation network. One thread per (b, l).
// h = GELU(w1 . x + b1); om = w2 . h + b2; zero ch 1,3,5;
// offsets = om[0,2,4]*(2/L); mod = softmax(om[3:36])[0:3].
// resize_linear is identity here (L_OUT == L_IN, align_corners).
// Emits pos p(b,k,l) = (base + kp + off + 1)*0.5*(L-1) and mod(b,k,l).
// ---------------------------------------------------------------------------
__global__ void dc1d_prep_offsets(const float* __restrict__ x,
                                  const float* __restrict__ w1,
                                  const float* __restrict__ b1,
                                  const float* __restrict__ w2,
                                  const float* __restrict__ b2,
                                  float* __restrict__ posw,
                                  float* __restrict__ modw) {
    __shared__ float sw1[4096];   // (4,16,64)
    __shared__ float sw2[576];    // (4,9,16)
    __shared__ float sb1[64];
    __shared__ float sb2[36];
    const int tid = threadIdx.x;
    for (int i = tid; i < 4096; i += 256) sw1[i] = w1[i];
    for (int i = tid; i < 576;  i += 256) sw2[i] = w2[i];
    if (tid < 64) sb1[tid] = b1[tid];
    if (tid < 36) sb2[tid] = b2[tid];
    __syncthreads();

    const int l = blockIdx.x * 256 + tid;
    const int b = blockIdx.y;
    const float* xb = x + (size_t)b * NCH * LSEQ;

    float h[64];
#pragma unroll
    for (int o = 0; o < 64; ++o) h[o] = sb1[o];

#pragma unroll
    for (int g = 0; g < 4; ++g) {
        for (int cg = 0; cg < 64; ++cg) {
            float xv = xb[(size_t)(g * 64 + cg) * LSEQ + l];
#pragma unroll
            for (int o = 0; o < 16; ++o)
                h[g * 16 + o] = fmaf(sw1[(g * 16 + o) * 64 + cg], xv, h[g * 16 + o]);
        }
    }
    // exact GELU
#pragma unroll
    for (int o = 0; o < 64; ++o)
        h[o] = 0.5f * h[o] * (1.0f + erff(h[o] * 0.70710678118654752f));

    float om[36];
#pragma unroll
    for (int g = 0; g < 4; ++g)
#pragma unroll
        for (int j = 0; j < 9; ++j) {
            float acc = sb2[g * 9 + j];
#pragma unroll
            for (int o = 0; o < 16; ++o)
                acc = fmaf(sw2[(g * 9 + j) * 16 + o], h[g * 16 + o], acc);
            om[g * 9 + j] = acc;
        }
    om[1] = 0.0f; om[3] = 0.0f; om[5] = 0.0f;

    // softmax over channels 3..35, keep first 3
    float mx = om[3];
#pragma unroll
    for (int i = 4; i < 36; ++i) mx = fmaxf(mx, om[i]);
    float s = 0.0f;
#pragma unroll
    for (int i = 3; i < 36; ++i) s += expf(om[i] - mx);
    float inv_s = 1.0f / s;

    const float base = -1.0f + (float)l * (2.0f / (float)(LSEQ - 1));
#pragma unroll
    for (int k = 0; k < KTAP; ++k) {
        float off = om[2 * k] * (2.0f / (float)LSEQ);
        float kp  = -0.5f + 0.5f * (float)k;
        float p   = (base + kp + off + 1.0f) * 0.5f * (float)(LSEQ - 1);
        size_t idx = ((size_t)b * KTAP + k) * LSEQ + l;
        posw[idx] = p;
        modw[idx] = expf(om[3 + k] - mx) * inv_s;
    }
}

// ---------------------------------------------------------------------------
// Kernel 2: repack weight (O=256, C=256, K=3) f32 -> bf16 in the WMMA 16x32
// A-operand layout (ISA 7.12.2, 16-bit A 16x32): per lane 16 bf16 values.
// Reduction index R = k*256 + c split into 24 chunks of 32 (kk = 0..31).
//   lanes 0-15 (M=lane):    V0-3 = kk 0..7,  V4-7 = kk 16..23
//   lanes 16-31 (M=lane-16):V0-3 = kk 8..15, V4-7 = kk 24..31
// Wp[((mt*24+t)*32+lane)*16 + (v*2+h)]
// ---------------------------------------------------------------------------
__global__ void dc1d_repack_w(const float* __restrict__ weight,
                              unsigned short* __restrict__ Wp) {
    const int gid = blockIdx.x * 256 + threadIdx.x;
    if (gid >= 16 * NCHUNK * 32 * 16) return;
    const int vh   = gid & 15;
    const int v    = vh >> 1;
    const int hh   = vh & 1;
    const int lane = (gid >> 4) & 31;
    const int rest = gid >> 9;
    const int t    = rest % NCHUNK;
    const int mt   = rest / NCHUNK;
    const int M    = mt * 16 + (lane & 15);
    const int kk   = ((v >> 2) << 4) + ((lane >> 4) << 3) + ((v & 3) << 1) + hh;
    const int k    = t >> 3;
    const int c    = ((t & 7) << 5) + kk;
    Wp[gid] = f32_to_bf16_rne(weight[((size_t)M * NCH + c) * KTAP + k]);
}

// ---------------------------------------------------------------------------
// Kernel 3: main deformable GEMM with bf16 WMMA, ping-pong LDS B tiles.
// Block = (b, 32-position tile). 256 threads = 8 waves; wave w owns M-tiles
// 2w,2w+1 (all 256 out-channels) x 2 N-tiles. Per chunk t:
//   issue A loads (chunk t) -> gather/lerp chunk t+1 into buf[(t+1)&1]
//   -> read B buf[t&1] -> 4 WMMAs -> single barrier.
// ---------------------------------------------------------------------------
__global__ void dc1d_deform_gemm(const float* __restrict__ x,
                                 const unsigned short* __restrict__ Wp,
                                 const float* __restrict__ bias,
                                 const float* __restrict__ posw,
                                 const float* __restrict__ modw,
                                 float* __restrict__ out) {
    __shared__ __align__(32) unsigned short ldsB[2][2][512]; // buf x tile x (32 lanes x 16 bf16)
    __shared__ int   s_i0[KTAP][32];
    __shared__ int   s_i1[KTAP][32];
    __shared__ float s_a0[KTAP][32];
    __shared__ float s_a1[KTAP][32];

    const int tid = threadIdx.x;
    const int b   = blockIdx.y;
    const int l0  = blockIdx.x * 32;

    if (tid < KTAP * 32) {
        int k = tid >> 5, n = tid & 31;
        size_t idx = ((size_t)b * KTAP + k) * LSEQ + l0 + n;
        float p  = posw[idx];
        float m  = modw[idx];
        float fi = floorf(p);
        int   i0 = (int)fi;
        float w  = p - fi;
        int   i1 = i0 + 1;
        s_a0[k][n] = (i0 >= 0 && i0 < LSEQ) ? (1.0f - w) * m : 0.0f;
        s_a1[k][n] = (i1 >= 0 && i1 < LSEQ) ? w * m : 0.0f;
        s_i0[k][n] = min(max(i0, 0), LSEQ - 1);
        s_i1[k][n] = min(max(i1, 0), LSEQ - 1);
    }
    __syncthreads();

    const int lane = tid & 31;
    const int wv   = tid >> 5;
    const int mt0  = wv * 2;
    const int mt1  = wv * 2 + 1;
    const int r0   = tid >> 4;   // 0..15
    const int n0   = tid & 15;   // 0..15

    const float* xb = x + (size_t)b * NCH * LSEQ;

    // gather+lerp chunk t into ping-pong buffer bb
    auto build = [&](int t, int bb) {
        const int k  = t >> 3;
        const int c0 = (t & 7) << 5;
#pragma unroll
        for (int m = 0; m < 2; ++m) {
            const int n  = n0 + 16 * m;
            const int i0 = s_i0[k][n], i1 = s_i1[k][n];
            const float a0c = s_a0[k][n], a1c = s_a1[k][n];
#pragma unroll
            for (int half = 0; half < 2; ++half) {
                int r = r0 + half * 16;
                const float* xc = xb + (size_t)(c0 + r) * LSEQ;
                float sv = xc[i0] * a0c + xc[i1] * a1c;  // lerp * mod (validity folded)
                int lw = n0 + (((r >> 3) & 1) << 4);      // dest lane
                ldsB[bb][m][lw * 16 + (r & 7) + ((r >> 4) << 3)] = f32_to_bf16_rne(sv);
            }
        }
    };

    v8f acc00 = {}; v8f acc01 = {};
    v8f acc10 = {}; v8f acc11 = {};

    build(0, 0);
    __syncthreads();

    for (int t = 0; t < NCHUNK; ++t) {
        const int bb = t & 1;

        // A operands for this chunk: issue early so they fly during the gather
        v16bf a0v = *(const v16bf*)(Wp + ((size_t)(mt0 * NCHUNK + t) * 32 + lane) * 16);
        v16bf a1v = *(const v16bf*)(Wp + ((size_t)(mt1 * NCHUNK + t) * 32 + lane) * 16);

        if (t + 1 < NCHUNK) {
            __builtin_prefetch(Wp + ((size_t)(mt0 * NCHUNK + t + 1) * 32 + lane) * 16, 0, 1);
            __builtin_prefetch(Wp + ((size_t)(mt1 * NCHUNK + t + 1) * 32 + lane) * 16, 0, 1);
            build(t + 1, bb ^ 1);
        }

        v16bf bv0 = *(const v16bf*)&ldsB[bb][0][lane * 16];
        v16bf bv1 = *(const v16bf*)&ldsB[bb][1][lane * 16];
        acc00 = __builtin_amdgcn_wmma_f32_16x16x32_bf16(false, a0v, false, bv0,
                                                        (short)0, acc00, false, false);
        acc01 = __builtin_amdgcn_wmma_f32_16x16x32_bf16(false, a0v, false, bv1,
                                                        (short)0, acc01, false, false);
        acc10 = __builtin_amdgcn_wmma_f32_16x16x32_bf16(false, a1v, false, bv0,
                                                        (short)0, acc10, false, false);
        acc11 = __builtin_amdgcn_wmma_f32_16x16x32_bf16(false, a1v, false, bv1,
                                                        (short)0, acc11, false, false);
        __syncthreads();  // publishes buf[bb^1]; protects buf[bb] until t+2
    }

    // C/D layout: N = lane&15; VGPR j -> M = j (lanes 0-15) / 8+j (lanes 16-31)
    const int nn = lane & 15;
    const int mo = (lane >> 4) << 3;
#pragma unroll
    for (int j = 0; j < 8; ++j) {
        int row0 = mt0 * 16 + mo + j;
        int row1 = mt1 * 16 + mo + j;
        float bz0 = bias[row0];
        float bz1 = bias[row1];
        size_t o0 = ((size_t)b * NCH + row0) * LSEQ + l0 + nn;
        size_t o1 = ((size_t)b * NCH + row1) * LSEQ + l0 + nn;
        out[o0]      = acc00[j] + bz0;
        out[o0 + 16] = acc01[j] + bz0;
        out[o1]      = acc10[j] + bz1;
        out[o1 + 16] = acc11[j] + bz1;
    }
}

// ---------------------------------------------------------------------------
extern "C" void kernel_launch(void* const* d_in, const int* in_sizes, int n_in,
                              void* d_out, int out_size, void* d_ws, size_t ws_size,
                              hipStream_t stream) {
    const float* x      = (const float*)d_in[0];
    const float* weight = (const float*)d_in[1];
    const float* bias   = (const float*)d_in[2];
    const float* w1     = (const float*)d_in[3];
    const float* b1     = (const float*)d_in[4];
    const float* w2     = (const float*)d_in[5];
    const float* b2     = (const float*)d_in[6];
    float* out = (float*)d_out;

    // workspace: Wp (393216 B) | posw (393216 B) | modw (393216 B)
    char* ws = (char*)d_ws;
    unsigned short* Wp   = (unsigned short*)ws;
    float*          posw = (float*)(ws + 393216);
    float*          modw = (float*)(ws + 786432);

    dc1d_prep_offsets<<<dim3(LSEQ / 256, NB), 256, 0, stream>>>(
        x, w1, b1, w2, b2, posw, modw);
    dc1d_repack_w<<<dim3((16 * NCHUNK * 32 * 16) / 256), 256, 0, stream>>>(
        weight, Wp);
    dc1d_deform_gemm<<<dim3(LSEQ / 32, NB), 256, 0, stream>>>(
        x, Wp, bias, posw, modw, out);
}
